// CausalAttention_32487132627054
// MI455X (gfx1250) — matrix operand order repaired and checked by reading, not compiled, not run on hardware
//
#include <hip/hip_runtime.h>
#include <hip/hip_bf16.h>
#include <stdint.h>

#define DEVI __device__ __forceinline__

typedef __bf16 v16bf __attribute__((ext_vector_type(16)));
typedef float  v8f   __attribute__((ext_vector_type(8)));

constexpr int Bb   = 2;
constexpr int Nn   = 2048;
constexpr int Dd   = 2048;
constexpr int Hh   = 16;
constexpr int HD   = 128;           // head dim
constexpr int Rtot = Bb * Nn;       // 4096 rows of x
constexpr int QKVN = 3 * Dd;        // 6144

struct alignas(16) U128 { unsigned long long a, b; };
struct alignas(16) F128 { float x, y, z, w; };
struct alignas(8)  U64  { unsigned int lo, hi; };
union Frag { U128 u[2]; v16bf v; };

DEVI unsigned short f2bf(float f) {
    unsigned int u = __builtin_bit_cast(unsigned int, f);
    unsigned int r = u + 0x7FFFu + ((u >> 16) & 1u);
    return (unsigned short)(r >> 16);
}

DEVI v8f vzero() { v8f z; for (int i = 0; i < 8; ++i) z[i] = 0.f; return z; }

// ---- CDNA5 async global->LDS copy (ASYNCcnt-tracked), guarded -------------
#if __has_builtin(__builtin_amdgcn_global_load_async_to_lds_b128)
#define HAVE_ASYNC_LDS 1
typedef int v4i_t __attribute__((vector_size(16)));
typedef __attribute__((address_space(1))) v4i_t* as1_v4i;
typedef __attribute__((address_space(3))) v4i_t* as3_v4i;
DEVI void async_cp16(const void* gsrc, void* ldst) {
    __builtin_amdgcn_global_load_async_to_lds_b128(
        (as1_v4i)(uintptr_t)gsrc, (as3_v4i)(uintptr_t)ldst, 0, 0);
}
#if __has_builtin(__builtin_amdgcn_s_wait_asynccnt)
#define WAIT_ASYNC() __builtin_amdgcn_s_wait_asynccnt(0)
#else
#define WAIT_ASYNC() asm volatile("s_wait_asynccnt 0" ::: "memory")
#endif
#else
#define HAVE_ASYNC_LDS 0
#define WAIT_ASYNC() ((void)0)
DEVI void async_cp16(const void* gsrc, void* ldst) {
    *(U128*)ldst = *(const U128*)gsrc;
}
#endif

// ---------------------------------------------------------------------------
// Tiled GEMM with bias: C[M,N] = A[M,K] @ W[K,N] + bias[N]
// Block: 256 threads (8 waves), tile 128x128, K-step 32.
// ---------------------------------------------------------------------------
template <bool A_BF16, bool OUT_BF16>
__global__ __launch_bounds__(256) void gemm_bias(
    const void* __restrict__ Aptr, const float* __restrict__ W,
    const float* __restrict__ bias, void* __restrict__ Cptr,
    int M, int N, int K)
{
    __shared__ __align__(16) unsigned short As[128][40]; // [m][k]
    __shared__ __align__(16) unsigned short Bs[128][40]; // [n][k] transposed

    const int tid  = threadIdx.x;
    const int lane = tid & 31;
    const int wave = tid >> 5;
    const int g    = lane >> 4;
    const int ln   = lane & 15;

    const int bm = blockIdx.y * 128;
    const int bn = blockIdx.x * 128;
    const int wm = (wave >> 2) * 64;     // 2 waves along M
    const int wn = (wave & 3) * 32;      // 4 waves along N

    v8f acc[4][2];
    for (int i = 0; i < 4; ++i)
        for (int j = 0; j < 2; ++j) acc[i][j] = vzero();

    for (int k0 = 0; k0 < K; k0 += 32) {
        // ---- stage A tile 128x32 ----
        if (A_BF16) {
            // raw bf16 copy: 512 x 16B chunks, 2 per thread (async if available)
            for (int i = 0; i < 2; ++i) {
                int c = tid + i * 256;
                int r = c >> 2, q = (c & 3) * 8;
                const unsigned short* src =
                    (const unsigned short*)Aptr + (size_t)(bm + r) * K + k0 + q;
                async_cp16(src, &As[r][q]);
            }
        } else {
            // fp32 -> bf16 convert, 16B vector loads, 8B LDS stores
            for (int i = 0; i < 4; ++i) {
                int c = tid + i * 256;
                int r = c >> 3, q = (c & 7) * 4;
                F128 fv = *(const F128*)((const float*)Aptr + (size_t)(bm + r) * K + k0 + q);
                U64 pk;
                pk.lo = (unsigned)f2bf(fv.x) | ((unsigned)f2bf(fv.y) << 16);
                pk.hi = (unsigned)f2bf(fv.z) | ((unsigned)f2bf(fv.w) << 16);
                *(U64*)&As[r][q] = pk;
            }
        }
        // ---- stage B tile 32x128 transposed to [n][k] ----
        for (int i = 0; i < 4; ++i) {
            int c = tid + i * 256;
            int kk = c >> 5, n0 = (c & 31) * 4;
            F128 fv = *(const F128*)(W + (size_t)(k0 + kk) * N + bn + n0);
            Bs[n0 + 0][kk] = f2bf(fv.x);
            Bs[n0 + 1][kk] = f2bf(fv.y);
            Bs[n0 + 2][kk] = f2bf(fv.z);
            Bs[n0 + 3][kk] = f2bf(fv.w);
        }
        if (A_BF16) WAIT_ASYNC();
        __syncthreads();

        Frag bf[2];
        for (int j = 0; j < 2; ++j) {
            const unsigned short* p = &Bs[wn + j * 16 + ln][g * 8];
            bf[j].u[0] = *(const U128*)p;
            bf[j].u[1] = *(const U128*)(p + 16);
        }
        for (int i = 0; i < 4; ++i) {
            Frag af;
            const unsigned short* p = &As[wm + i * 16 + ln][g * 8];
            af.u[0] = *(const U128*)p;
            af.u[1] = *(const U128*)(p + 16);
            for (int j = 0; j < 2; ++j)
                acc[i][j] = __builtin_amdgcn_wmma_f32_16x16x32_bf16(
                    false, af.v, false, bf[j].v, (short)0, acc[i][j], false, false);
        }
        __syncthreads();
    }

    // epilogue
    for (int i = 0; i < 4; ++i)
        for (int j = 0; j < 2; ++j) {
            int col = bn + wn + j * 16 + ln;
            float bv = bias[col];
            for (int r = 0; r < 8; ++r) {
                int row = bm + wm + i * 16 + r + g * 8;
                float val = acc[i][j][r] + bv;
                if (OUT_BF16)
                    ((unsigned short*)Cptr)[(size_t)row * N + col] = f2bf(val);
                else
                    ((float*)Cptr)[(size_t)row * N + col] = val;
            }
        }
}

// ---------------------------------------------------------------------------
// Fused attention: per (b,h), 128 query rows per block (16 per wave).
// Online (flash) softmax over 2048 keys, mask*sigmoid applied to scores.
// qkv: bf16 [4096][6144] (q|k|v columns). Output bf16 [4096][2048].
// ---------------------------------------------------------------------------
__global__ __launch_bounds__(256) void attention_kernel(
    const unsigned short* __restrict__ qkv,
    const float* __restrict__ mask,
    unsigned short* __restrict__ attn_out)
{
    __shared__ __align__(16) unsigned short Ks[32][136];   // K rows j..j+31 x hd128
    __shared__ __align__(16) unsigned short Vt[128][40];   // V transposed [hd][row]
    __shared__ __align__(16) unsigned short Pl[8][16][32]; // per-wave P scratch
    __shared__ __align__(16) float          Ms[128][36];   // mask tile [qrow][32 cols]

    const int tid  = threadIdx.x;
    const int lane = tid & 31;
    const int wave = tid >> 5;
    const int g    = lane >> 4;
    const int ln   = lane & 15;

    const int bh = blockIdx.z;          // b*H + h
    const int b  = bh >> 4, h = bh & 15;
    const int qt0   = blockIdx.y * 128;              // block's first query row
    const int qrow0 = qt0 + wave * 16;               // within-sequence row
    const int grow0 = b * Nn + qrow0;                // global row in qkv

    const float scale = 0.08838834764831845f;        // 1/sqrt(128)

    // preload Q fragments (hd=128 -> 4 K-chunks of 32)
    Frag qf[4];
    {
        const unsigned short* qbase = qkv + (size_t)(grow0 + ln) * QKVN + h * HD;
        for (int c = 0; c < 4; ++c) {
            const unsigned short* p = qbase + c * 32 + g * 8;
            qf[c].u[0] = *(const U128*)p;
            qf[c].u[1] = *(const U128*)(p + 16);
        }
    }

    v8f acco[8];
    for (int t = 0; t < 8; ++t) acco[t] = vzero();
    float mrow[8], lrow[8];
    for (int r = 0; r < 8; ++r) { mrow[r] = -1e30f; lrow[r] = 0.f; }

    for (int j = 0; j < Nn; j += 32) {
        // ---- stage K tile (raw bf16 copy, async): 32 rows x 128 ----
        {
            int r = tid >> 3, c = (tid & 7) * 16;
            const unsigned short* src =
                qkv + (size_t)(b * Nn + j + r) * QKVN + Dd + h * HD + c;
            async_cp16(src,     &Ks[r][c]);
            async_cp16(src + 8, &Ks[r][c + 8]);
            if (j + 32 < Nn)
                __builtin_prefetch(src + (size_t)32 * QKVN, 0, 1);
        }
        // ---- stage mask tile (raw f32 copy, async): 128 rows x 32 cols ----
        for (int i = 0; i < 4; ++i) {
            int c = tid + i * 256;
            int row = c >> 3, c4 = (c & 7) * 4;
            const float* msrc = mask + ((size_t)h * Nn + qt0 + row) * Nn + j + c4;
            async_cp16(msrc, &Ms[row][c4]);
        }
        // ---- stage V tile transposed: Vt[hd][row] ----
        {
            int r = tid >> 3, c0 = (tid & 7) * 16;
            const unsigned short* src =
                qkv + (size_t)(b * Nn + j + r) * QKVN + 2 * Dd + h * HD + c0;
            for (int c = 0; c < 16; ++c) Vt[c0 + c][r] = src[c];
            if (j + 32 < Nn)
                __builtin_prefetch(src + (size_t)32 * QKVN, 0, 1);
        }
        WAIT_ASYNC();
        __syncthreads();

        // ---- S = Q @ K^T for 2 column tiles of 16 ----
        v8f s[2];
        for (int t = 0; t < 2; ++t) {
            v8f sv = vzero();
            for (int c = 0; c < 4; ++c) {
                Frag kb;
                const unsigned short* p = &Ks[t * 16 + ln][c * 32 + g * 8];
                kb.u[0] = *(const U128*)p;
                kb.u[1] = *(const U128*)(p + 16);
                sv = __builtin_amdgcn_wmma_f32_16x16x32_bf16(
                    false, qf[c].v, false, kb.v, (short)0, sv, false, false);
            }
            s[t] = sv;
        }

        // ---- scale + multiplicative sigmoid mask (from LDS) ----
        for (int t = 0; t < 2; ++t) {
            for (int r = 0; r < 8; ++r) {
                float mk = Ms[wave * 16 + g * 8 + r][t * 16 + ln];
                float sg = __builtin_amdgcn_rcpf(1.f + __expf(-mk));
                s[t][r] = s[t][r] * (scale * sg);
            }
        }

        // ---- online softmax (rows live in 16-lane halves) ----
        float pex[2][8];
        for (int r = 0; r < 8; ++r) {
            float tm = fmaxf(s[0][r], s[1][r]);
            for (int off = 8; off >= 1; off >>= 1)
                tm = fmaxf(tm, __shfl_xor(tm, off, 32));
            float mn   = fmaxf(mrow[r], tm);
            float corr = __expf(mrow[r] - mn);
            float p0   = __expf(s[0][r] - mn);
            float p1   = __expf(s[1][r] - mn);
            float ts   = p0 + p1;
            for (int off = 8; off >= 1; off >>= 1)
                ts += __shfl_xor(ts, off, 32);
            lrow[r] = lrow[r] * corr + ts;
            mrow[r] = mn;
            for (int t = 0; t < 8; ++t) acco[t][r] *= corr;
            pex[0][r] = p0; pex[1][r] = p1;
        }

        // ---- re-layout P (C -> A layout) via per-wave LDS, then P @ V ----
        {
            unsigned short (&P)[16][32] = Pl[wave];
            for (int t = 0; t < 2; ++t)
                for (int r = 0; r < 8; ++r)
                    P[r + g * 8][t * 16 + ln] = f2bf(pex[t][r]);
            asm volatile("s_wait_dscnt 0" ::: "memory"); // intra-wave cross-lane LDS hazard

            Frag pf;
            const unsigned short* pp = &P[ln][g * 8];
            pf.u[0] = *(const U128*)pp;
            pf.u[1] = *(const U128*)(pp + 16);

            for (int t = 0; t < 8; ++t) {
                Frag vf;
                const unsigned short* vp = &Vt[t * 16 + ln][g * 8];
                vf.u[0] = *(const U128*)vp;
                vf.u[1] = *(const U128*)(vp + 16);
                acco[t] = __builtin_amdgcn_wmma_f32_16x16x32_bf16(
                    false, pf.v, false, vf.v, (short)0, acco[t], false, false);
            }
        }
        __syncthreads();
    }

    // epilogue: normalize by row sum, write bf16
    for (int r = 0; r < 8; ++r) {
        float inv = __builtin_amdgcn_rcpf(lrow[r]);
        int row = grow0 + r + g * 8;
        for (int t = 0; t < 8; ++t) {
            int col = h * HD + t * 16 + ln;
            attn_out[(size_t)row * Dd + col] = f2bf(acco[t][r] * inv);
        }
    }
}

// ---------------------------------------------------------------------------
extern "C" void kernel_launch(void* const* d_in, const int* in_sizes, int n_in,
                              void* d_out, int out_size, void* d_ws, size_t ws_size,
                              hipStream_t stream) {
    const float* x    = (const float*)d_in[0];
    const float* Wqkv = (const float*)d_in[1];
    const float* bqkv = (const float*)d_in[2];
    const float* Wout = (const float*)d_in[3];
    const float* bout = (const float*)d_in[4];
    const float* mask = (const float*)d_in[5];
    float* out = (float*)d_out;

    unsigned short* qkv_bf  = (unsigned short*)d_ws;               // 48 MB
    unsigned short* attn_bf = qkv_bf + (size_t)Rtot * QKVN;        // +16 MB

    dim3 blk(256);

    // 1) qkv = x @ W_qkv + b_qkv   (fp32 in -> bf16 out)
    gemm_bias<false, true><<<dim3(QKVN / 128, Rtot / 128), blk, 0, stream>>>(
        (const void*)x, Wqkv, bqkv, (void*)qkv_bf, Rtot, QKVN, Dd);

    // 2) fused masked attention (flash-style)
    attention_kernel<<<dim3(1, Nn / 128, Bb * Hh), blk, 0, stream>>>(
        qkv_bf, mask, attn_bf);

    // 3) out = attn @ W_out + b_out   (bf16 in -> fp32 out)
    gemm_bias<true, false><<<dim3(Dd / 128, Rtot / 128), blk, 0, stream>>>(
        (const void*)attn_bf, Wout, bout, (void*)out, Rtot, Dd, Dd);
}
